// GNN_EBM_54073638256943
// MI455X (gfx1250) — matrix operand order repaired and checked by reading, not compiled, not run on hardware
//
#include <hip/hip_runtime.h>
#include <hip/hip_bf16.h>

typedef __attribute__((ext_vector_type(16))) _Float16 v16h;
typedef __attribute__((ext_vector_type(8)))  _Float16 v8h;
typedef __attribute__((ext_vector_type(8)))  float    v8f;

#define NXg 96
#define NYg 96
#define NNODE (NXg*NYg)        // 9216
#define NBATCH 8
#define NROWS (NBATCH*NNODE)   // 73728

__device__ __forceinline__ float gelu_exact(float y) {
    return 0.5f * y * (1.0f + erff(y * 0.70710678118654752f));
}

__device__ __forceinline__ float wave_sum(float v) {
#pragma unroll
    for (int m = 16; m >= 1; m >>= 1) v += __shfl_xor(v, m, 32);
    return v;
}

// ---------------------------------------------------------------------------
// Repack f32 weight matrix (K x DOUT, row-major) into the per-lane fragment
// order of V_WMMA_F32_16X16X32_F16's B operand (32x16 tile):
//   lane l holds column n = t*16 + (l&15); half h holds K = kt*32 + (h>>3)*16 + (l>>4)*8 + (h&7)
// Packed index: (( t*KT + kt )*32 + lane)*16 + h  -> one contiguous v16h/lane.
// ---------------------------------------------------------------------------
__global__ void pack_w_kernel(const float* __restrict__ W, _Float16* __restrict__ Wpk,
                              int KT, int DOUT, int total) {
    int o = blockIdx.x * blockDim.x + threadIdx.x;
    if (o >= total) return;
    int h    = o & 15;
    int lane = (o >> 4) & 31;
    int p    = o >> 9;
    int kt   = p % KT;
    int t    = p / KT;
    int n = t * 16 + (lane & 15);
    int k = kt * 32 + (h >> 3) * 16 + (lane >> 4) * 8 + (h & 7);
    Wpk[o] = (_Float16)W[k * DOUT + n];
}

// ---------------------------------------------------------------------------
// Node encoder: h = gelu(LN64(concat(u,x) @ enc_w + enc_b)). One wave per node.
// ---------------------------------------------------------------------------
__global__ void enc_kernel(const float* __restrict__ u, const float* __restrict__ x,
                           const float* __restrict__ W, const float* __restrict__ bvec,
                           const float* __restrict__ g, const float* __restrict__ be,
                           _Float16* __restrict__ hout) {
    int wave = threadIdx.x >> 5, lane = threadIdx.x & 31;
    int row = blockIdx.x * 8 + wave;
    float f0 = u[row];
    float f1 = x[row * 3 + 0], f2 = x[row * 3 + 1], f3 = x[row * 3 + 2];
    float z[2]; float s1 = 0.f, s2 = 0.f;
#pragma unroll
    for (int f = 0; f < 2; ++f) {
        int d = lane + f * 32;
        float t = f0 * W[d] + f1 * W[64 + d] + f2 * W[128 + d] + f3 * W[192 + d] + bvec[d];
        z[f] = t; s1 += t; s2 += t * t;
    }
    s1 = wave_sum(s1); s2 = wave_sum(s2);
    float mean = s1 * (1.0f / 64.0f);
    float var  = s2 * (1.0f / 64.0f) - mean * mean;
    float rstd = rsqrtf(var + 1e-5f);
#pragma unroll
    for (int f = 0; f < 2; ++f) {
        int d = lane + f * 32;
        float y = (z[f] - mean) * rstd * g[d] + be[d];
        hout[(size_t)row * 64 + d] = (_Float16)gelu_exact(y);
    }
}

// ---------------------------------------------------------------------------
// Dual GEMM via WMMA: Oself = X@Wself + bself ; Onb = X@Wnb + bnb (f32 out).
// X is f16 row-major (NROWS x DIN). One 16-row M-tile per block; 2 waves split
// the DOUT/16 N-tiles. A fragments loaded per the ISA 16-bit A layout:
//   lane (l&15)=M row; halfs 0..7 -> K = kt*32 + (l>>4)*8 + 0..7,
//   halfs 8..15 -> +16. Two b128 loads per fragment, held in registers.
// ---------------------------------------------------------------------------
template <int DIN, int DOUT>
__global__ void gemm_dual_kernel(const _Float16* __restrict__ X,
                                 const _Float16* __restrict__ Wself, const float* __restrict__ bself,
                                 const _Float16* __restrict__ Wnb,   const float* __restrict__ bnb,
                                 float* __restrict__ Oself, float* __restrict__ Onb) {
    constexpr int KT  = DIN / 32;   // K-tiles of 32
    constexpr int NT  = DOUT / 16;  // N-tiles of 16
    constexpr int TPW = NT / 2;     // tiles per wave (2 waves per block)
    const int wave = threadIdx.x >> 5;
    const int lane = threadIdx.x & 31;
    const int mrow = blockIdx.x * 16 + (lane & 15);
    const int hi   = lane >> 4;

    v16h a[KT];
#pragma unroll
    for (int kt = 0; kt < KT; ++kt) {
        const _Float16* p = X + (size_t)mrow * DIN + kt * 32 + hi * 8;
        v8h lo  = *(const v8h*)(p);
        v8h hi8 = *(const v8h*)(p + 16);
#pragma unroll
        for (int i = 0; i < 8; ++i) { a[kt][i] = lo[i]; a[kt][8 + i] = hi8[i]; }
    }

    const v16h* Wf[2]  = {(const v16h*)Wself, (const v16h*)Wnb};
    const float* bb[2] = {bself, bnb};
    float* oo[2]       = {Oself, Onb};

#pragma unroll
    for (int m = 0; m < 2; ++m) {
#pragma unroll
        for (int ti = 0; ti < TPW; ++ti) {
            const int t = wave * TPW + ti;
            const float bv = bb[m][t * 16 + (lane & 15)];
            v8f c;
#pragma unroll
            for (int r = 0; r < 8; ++r) c[r] = bv;
#pragma unroll
            for (int kt = 0; kt < KT; ++kt) {
                v16h bfrag = Wf[m][((size_t)t * KT + kt) * 32 + lane];
                c = __builtin_amdgcn_wmma_f32_16x16x32_f16(
                        false, a[kt], false, bfrag, (short)0, c, false, false);
            }
            // D layout: VGPR r -> M = r + hi*8, N = lane&15
            float* outp = oo[m] + (size_t)(blockIdx.x * 16) * DOUT + t * 16 + (lane & 15);
#pragma unroll
            for (int r = 0; r < 8; ++r) {
                int mm = r + hi * 8;
                outp[(size_t)mm * DOUT] = c[r];
            }
        }
    }
}

// ---------------------------------------------------------------------------
// Fused: z = h_self + stencil(h_nb)/(deg+1e-8); LayerNorm(DOUT); GELU; -> f16.
// The dense row-normalized grid adjacency IS this 5-point stencil.
// One wave per node; wave32 shfl reductions for mean/var.
// ---------------------------------------------------------------------------
template <int DOUT>
__global__ void agg_ln_gelu_kernel(const float* __restrict__ Self, const float* __restrict__ Nb,
                                   const float* __restrict__ g, const float* __restrict__ be,
                                   _Float16* __restrict__ hout) {
    constexpr int F = DOUT / 32;
    int wave = threadIdx.x >> 5, lane = threadIdx.x & 31;
    int idx = blockIdx.x * 8 + wave;
    int b = idx / NNODE;
    int n = idx - b * NNODE;
    int i = n / NYg, j = n - i * NYg;
    int deg = (i > 0) + (i < NXg - 1) + (j > 0) + (j < NYg - 1);
    float inv = 1.0f / ((float)deg + 1e-8f);
    const float* nbb = Nb + (size_t)b * NNODE * DOUT;

    float z[F]; float s1 = 0.f, s2 = 0.f;
#pragma unroll
    for (int f = 0; f < F; ++f) {
        int d = lane + f * 32;
        float acc = 0.f;
        if (i > 0)        acc += nbb[(size_t)(n - NYg) * DOUT + d];
        if (i < NXg - 1)  acc += nbb[(size_t)(n + NYg) * DOUT + d];
        if (j > 0)        acc += nbb[(size_t)(n - 1)   * DOUT + d];
        if (j < NYg - 1)  acc += nbb[(size_t)(n + 1)   * DOUT + d];
        float t = Self[(size_t)idx * DOUT + d] + acc * inv;
        z[f] = t; s1 += t; s2 += t * t;
    }
    s1 = wave_sum(s1); s2 = wave_sum(s2);
    float mean = s1 / (float)DOUT;
    float var  = s2 / (float)DOUT - mean * mean;
    float rstd = rsqrtf(var + 1e-5f);
#pragma unroll
    for (int f = 0; f < F; ++f) {
        int d = lane + f * 32;
        float y = (z[f] - mean) * rstd * g[d] + be[d];
        hout[(size_t)idx * DOUT + d] = (_Float16)gelu_exact(y);
    }
}

// ---------------------------------------------------------------------------
// Global mean pool over N per batch (f16 in, f32 out).
// ---------------------------------------------------------------------------
__global__ void pool_kernel(const _Float16* __restrict__ h, float* __restrict__ gf) {
    int b = blockIdx.x, d = threadIdx.x;
    const _Float16* p = h + (size_t)b * NNODE * 64 + d;
    float acc = 0.f;
    for (int n = 0; n < NNODE; ++n) acc += (float)p[(size_t)n * 64];
    gf[b * 64 + d] = acc * (1.0f / (float)NNODE);
}

// ---------------------------------------------------------------------------
// Energy head: e = gelu(LN32(gf@h1_w + h1_b)) @ h2_w + h2_b. One wave/batch.
// ---------------------------------------------------------------------------
__global__ void head_kernel(const float* __restrict__ gf,
                            const float* __restrict__ w1, const float* __restrict__ b1,
                            const float* __restrict__ g1, const float* __restrict__ be1,
                            const float* __restrict__ w2, const float* __restrict__ b2,
                            float* __restrict__ out) {
    int b = threadIdx.x >> 5, lane = threadIdx.x & 31;
    float t = b1[lane];
    for (int k = 0; k < 64; ++k) t += gf[b * 64 + k] * w1[k * 32 + lane];
    float s1 = wave_sum(t), s2 = wave_sum(t * t);
    float mean = s1 * (1.0f / 32.0f);
    float var  = s2 * (1.0f / 32.0f) - mean * mean;
    float rstd = rsqrtf(var + 1e-5f);
    float y = (t - mean) * rstd * g1[lane] + be1[lane];
    float e = wave_sum(gelu_exact(y) * w2[lane]);
    if (lane == 0) out[b] = e + b2[0];
}

// ---------------------------------------------------------------------------
extern "C" void kernel_launch(void* const* d_in, const int* in_sizes, int n_in,
                              void* d_out, int out_size, void* d_ws, size_t ws_size,
                              hipStream_t stream) {
    (void)in_sizes; (void)n_in; (void)out_size; (void)ws_size;

    const float* u        = (const float*)d_in[0];
    const float* x        = (const float*)d_in[1];
    // d_in[2] = adj : unused (replaced analytically by the 5-point stencil)
    const float* enc_w    = (const float*)d_in[3];
    const float* enc_b    = (const float*)d_in[4];
    const float* enc_g    = (const float*)d_in[5];
    const float* enc_beta = (const float*)d_in[6];
    // layer param blocks: s_w, s_b, n_w, n_b, ln_g, ln_b
    const int L1 = 7, L2 = 13, L3 = 19;
    const float* h1_w = (const float*)d_in[25];
    const float* h1_b = (const float*)d_in[26];
    const float* h1_g = (const float*)d_in[27];
    const float* h1_be= (const float*)d_in[28];
    const float* h2_w = (const float*)d_in[29];
    const float* h2_b = (const float*)d_in[30];
    float* out = (float*)d_out;

    // Workspace layout
    char* ws = (char*)d_ws;
    const size_t SZ_H16 = (size_t)NROWS * 128 * sizeof(_Float16);  // 18,874,368
    const size_t SZ_F32 = (size_t)NROWS * 128 * sizeof(float);     // 37,748,736
    _Float16* hA  = (_Float16*)(ws);
    _Float16* hB  = (_Float16*)(ws + SZ_H16);
    float*    Sf  = (float*)(ws + 2 * SZ_H16);
    float*    Nf  = (float*)(ws + 2 * SZ_H16 + SZ_F32);
    _Float16* wpk = (_Float16*)(ws + 2 * SZ_H16 + 2 * SZ_F32);
    float*    gf  = (float*)(ws + 2 * SZ_H16 + 2 * SZ_F32 + 65536 * sizeof(_Float16));

    // packed weight offsets (halfs): s1,n1 (64x128); s2,n2 (128x128); s3,n3 (128x64)
    _Float16* s1p = wpk + 0;
    _Float16* n1p = wpk + 8192;
    _Float16* s2p = wpk + 16384;
    _Float16* n2p = wpk + 32768;
    _Float16* s3p = wpk + 49152;
    _Float16* n3p = wpk + 57344;

    // --- repack weights into WMMA B-fragment order ---
    pack_w_kernel<<<(8192 + 255) / 256, 256, 0, stream>>>((const float*)d_in[L1 + 0], s1p, 2, 128, 8192);
    pack_w_kernel<<<(8192 + 255) / 256, 256, 0, stream>>>((const float*)d_in[L1 + 2], n1p, 2, 128, 8192);
    pack_w_kernel<<<(16384 + 255) / 256, 256, 0, stream>>>((const float*)d_in[L2 + 0], s2p, 4, 128, 16384);
    pack_w_kernel<<<(16384 + 255) / 256, 256, 0, stream>>>((const float*)d_in[L2 + 2], n2p, 4, 128, 16384);
    pack_w_kernel<<<(8192 + 255) / 256, 256, 0, stream>>>((const float*)d_in[L3 + 0], s3p, 4, 64, 8192);
    pack_w_kernel<<<(8192 + 255) / 256, 256, 0, stream>>>((const float*)d_in[L3 + 2], n3p, 4, 64, 8192);

    // --- encoder ---
    enc_kernel<<<NROWS / 8, 256, 0, stream>>>(u, x, enc_w, enc_b, enc_g, enc_beta, hA);

    const int GB = NROWS / 16;  // 4608 M-tiles

    // --- layer 1: 64 -> 128 ---
    gemm_dual_kernel<64, 128><<<GB, 64, 0, stream>>>(hA, s1p, (const float*)d_in[L1 + 1],
                                                     n1p, (const float*)d_in[L1 + 3], Sf, Nf);
    agg_ln_gelu_kernel<128><<<NROWS / 8, 256, 0, stream>>>(Sf, Nf, (const float*)d_in[L1 + 4],
                                                           (const float*)d_in[L1 + 5], hB);
    // --- layer 2: 128 -> 128 ---
    gemm_dual_kernel<128, 128><<<GB, 64, 0, stream>>>(hB, s2p, (const float*)d_in[L2 + 1],
                                                      n2p, (const float*)d_in[L2 + 3], Sf, Nf);
    agg_ln_gelu_kernel<128><<<NROWS / 8, 256, 0, stream>>>(Sf, Nf, (const float*)d_in[L2 + 4],
                                                           (const float*)d_in[L2 + 5], hA);
    // --- layer 3: 128 -> 64 ---
    gemm_dual_kernel<128, 64><<<GB, 64, 0, stream>>>(hA, s3p, (const float*)d_in[L3 + 1],
                                                     n3p, (const float*)d_in[L3 + 3], Sf, Nf);
    agg_ln_gelu_kernel<64><<<NROWS / 8, 256, 0, stream>>>(Sf, Nf, (const float*)d_in[L3 + 4],
                                                          (const float*)d_in[L3 + 5], hB);

    // --- pool + head ---
    pool_kernel<<<NBATCH, 64, 0, stream>>>(hB, gf);
    head_kernel<<<1, 256, 0, stream>>>(gf, h1_w, h1_b, h1_g, h1_be, h2_w, h2_b, out);
}
